// S4DSSM_14199161880583
// MI455X (gfx1250) — matrix operand order repaired and checked by reading, not compiled, not run on hardware
//
#include <hip/hip_runtime.h>
#include <hip/hip_bf16.h>

// S4D SSM fused pipeline for MI455X (gfx1250, wave32, WMMA 16x16x32 f16->f32)
// B=8, T=4096, D=512, N=64. Bandwidth-bound (~128MB min HBM traffic ~5.5us
// at 23.3 TB/s); chunked scan (chunk=128 rows) exposes 256-way parallelism.
// LDS tiles are staged in WMMA *fragment order* so each lane's v16h fragment
// is a contiguous 32-byte LDS load (2x ds_load_b128), not 16 scalar u16s.

typedef __attribute__((ext_vector_type(16))) _Float16 v16h;
typedef __attribute__((ext_vector_type(4)))  _Float16 v4h;
typedef __attribute__((ext_vector_type(2)))  _Float16 v2h;
typedef __attribute__((ext_vector_type(8)))  float    v8f;

#define BSZ   8
#define TLEN  4096
#define DDIM  512
#define NDIM  64
#define BT    (BSZ * TLEN)      // 32768 rows
#define CHK   128               // rows per chunk (local scan length)
#define NCH   (BT / CHK)        // 256 chunks
#define CPB   (TLEN / CHK)      // 32 chunks per batch element

// K-permutation for A tiles: lanes 0-15 consume K{0-7,16-23}, lanes 16-31
// consume K{8-15,24-31}. Storing K at pos = (K&7)|((K&8)<<1)|((K&16)>>1)
// makes each lane group's 16 halves contiguous.
__device__ __forceinline__ int kperm(int k) {
    return (k & 7) | ((k & 8) << 1) | ((k & 16) >> 1);
}

__device__ __forceinline__ float decay_of(const float* log_lambda, const float* log_dt, int n) {
    float dt  = __expf(log_dt[0]);
    float lam = -__expf(log_lambda[n]);
    return __expf(lam * dt);
}

// ---- Kernel 1: Bu = x @ B^T for one 128-row chunk, then local scan --------
__global__ __launch_bounds__(256)
void gemm1_scan(const float* __restrict__ x, const float* __restrict__ Bm,
                const float* __restrict__ log_lambda, const float* __restrict__ log_dt,
                float* __restrict__ hs, float* __restrict__ Scarry) {
    __shared__ __align__(64) _Float16 aT[CHK][32];   // x tile, K-permuted    8 KB
    __shared__ __align__(64) _Float16 bF[NDIM][32];  // B tile [n][k]         4 KB
    __shared__ float bu[CHK][NDIM + 1];              // f32 accum + scan    ~33 KB

    const int tid   = threadIdx.x;
    const int lane  = tid & 31;
    const int wave  = tid >> 5;             // 0..7
    const int chunk = blockIdx.x;
    const int row0  = chunk * CHK;

    v8f acc[4] = {};                        // wave -> rows 16*wave..+15, 4 n-blocks

    for (int d0 = 0; d0 < DDIM; d0 += 32) {
        __syncthreads();
        // stage x tile: 128x32 f32 -> f16, K-permuted, v4h packed stores
        {
            int r  = tid >> 1;
            int k0 = (tid & 1) * 16;        // this thread covers K = k0..k0+15
            const float* src = x + (size_t)(row0 + r) * DDIM + d0 + k0;
            if (d0 + 32 < DDIM) __builtin_prefetch(src + 32, 0, 0);
#pragma unroll
            for (int seg = 0; seg < 2; ++seg) {         // 8 K's per segment
                int pos = kperm(k0 + seg * 8);          // contiguous 8-run
                v4h t0, t1;
#pragma unroll
                for (int i = 0; i < 4; ++i) t0[i] = (_Float16)src[seg * 8 + i];
#pragma unroll
                for (int i = 0; i < 4; ++i) t1[i] = (_Float16)src[seg * 8 + 4 + i];
                *(v4h*)&aT[r][pos]     = t0;
                *(v4h*)&aT[r][pos + 4] = t1;
            }
        }
        // stage B tile: bF[n][kperm(k)] = Bm[n*D + d0 + k] (K-permuted like aT)
        {
            int n  = tid >> 2;
            int k0 = (tid & 3) * 8;
            const float* src = Bm + (size_t)n * DDIM + d0 + k0;
            v4h t0, t1;
#pragma unroll
            for (int i = 0; i < 4; ++i) t0[i] = (_Float16)src[i];
#pragma unroll
            for (int i = 0; i < 4; ++i) t1[i] = (_Float16)src[4 + i];
            *(v4h*)&bF[n][kperm(k0)]     = t0;
            *(v4h*)&bF[n][kperm(k0) + 4] = t1;
        }
        __syncthreads();

        // fragments: single contiguous v16h per lane (2x ds_load_b128)
        v16h a = *(const v16h*)&aT[wave * 16 + (lane & 15)][(lane >> 4) * 16];
#pragma unroll
        for (int nb = 0; nb < 4; ++nb) {
            v16h b = *(const v16h*)&bF[nb * 16 + (lane & 15)][(lane >> 4) * 16];
            acc[nb] = __builtin_amdgcn_wmma_f32_16x16x32_f16(
                false, a, false, b, (short)0, acc[nb], false, false);
        }
    }
    __syncthreads();

    // spill accumulators (C/D layout: VGPR v -> M = v + 8*(lane>=16), N = lane&15)
    {
        int moff = (lane >> 4) << 3;
        int ncol = lane & 15;
#pragma unroll
        for (int nb = 0; nb < 4; ++nb)
#pragma unroll
            for (int v = 0; v < 8; ++v)
                bu[wave * 16 + moff + v][nb * 16 + ncol] = acc[nb][v];
    }
    __syncthreads();

    // chunk-local scan over 128 rows (one thread per n)
    if (tid < NDIM) {
        float dec = decay_of(log_lambda, log_dt, tid);
        float h = 0.0f;
#pragma unroll 4
        for (int t = 0; t < CHK; ++t) {
            h = h * dec + bu[t][tid];
            bu[t][tid] = h;
        }
        Scarry[(size_t)chunk * NDIM + tid] = h;
    }
    __syncthreads();

    // write local hs (f32) to workspace, coalesced
    {
        int n  = tid & 63;
        int rg = tid >> 6;                  // 0..3
#pragma unroll 4
        for (int i = 0; i < CHK / 4; ++i) {
            int r = rg + 4 * i;
            hs[(size_t)(row0 + r) * NDIM + n] = bu[r][n];
        }
    }
}

// ---- Kernel 2: sequential carry combine across chunks (per b, n) ----------
__global__ __launch_bounds__(512)
void carry_kernel(const float* __restrict__ S,
                  const float* __restrict__ log_lambda, const float* __restrict__ log_dt,
                  float* __restrict__ carryIn) {
    int t = threadIdx.x;                    // 512 = 8 b * 64 n
    int b = t >> 6, n = t & 63;
    float dt   = __expf(log_dt[0]);
    float lam  = -__expf(log_lambda[n]);
    float decL = __expf(lam * dt * (float)CHK);   // decay^CHK
    float P = 0.0f;
    for (int c = 0; c < CPB; ++c) {
        int idx = b * CPB + c;
        carryIn[(size_t)idx * NDIM + n] = P;      // state entering chunk c
        P = decL * P + S[(size_t)idx * NDIM + n];
    }
}

// ---- Kernel 3: hs fix-up + y = hs @ C^T -----------------------------------
__global__ __launch_bounds__(256)
void fixup_gemm2(const float* __restrict__ hs, const float* __restrict__ carryIn,
                 const float* __restrict__ Cm,
                 const float* __restrict__ log_lambda, const float* __restrict__ log_dt,
                 float* __restrict__ y) {
    __shared__ __align__(64) _Float16 hT[CHK][NDIM];  // hs tile, K-permuted  16 KB
    __shared__ __align__(64) _Float16 cF[128][NDIM];  // C^T tile [d'][n]     16 KB

    const int tid   = threadIdx.x;
    const int lane  = tid & 31;
    const int wave  = tid >> 5;
    const int chunk = blockIdx.x;
    const int row0  = chunk * CHK;
    const int d0    = blockIdx.y * 128;

    // fix-up + f16 staging: hT[t][perm(n)] = hs[t][n] + decay^(t+1)*carry
    // thread owns an n-pair over 16 rows -> packed v2h LDS stores
    {
        int n0 = (tid & 31) * 2;
        int r0 = (tid >> 5) * 16;
        int blk = n0 >> 5, wi = n0 & 31;
        int pos = blk * 32 + kperm(wi);     // pair stays adjacent under kperm
        float dec0   = decay_of(log_lambda, log_dt, n0);
        float dec1   = decay_of(log_lambda, log_dt, n0 + 1);
        float carry0 = carryIn[(size_t)chunk * NDIM + n0];
        float carry1 = carryIn[(size_t)chunk * NDIM + n0 + 1];
        float dp0 = __powf(dec0, (float)(r0 + 1));
        float dp1 = __powf(dec1, (float)(r0 + 1));
#pragma unroll 4
        for (int r = r0; r < r0 + 16; ++r) {
            const float* src = hs + (size_t)(row0 + r) * NDIM + n0;
            v2h p;
            p[0] = (_Float16)(src[0] + dp0 * carry0);
            p[1] = (_Float16)(src[1] + dp1 * carry1);
            *(v2h*)&hT[r][pos] = p;
            dp0 *= dec0; dp1 *= dec1;
        }
    }
    // stage C^T: cF[d'][blk*32+kperm(n&31)] = Cm[(d0+d')*N + n] (K-permuted like hT)
    {
        int dp = tid >> 1;
        int n0 = (tid & 1) * 32;
        const float* src = Cm + (size_t)(d0 + dp) * NDIM + n0;
#pragma unroll
        for (int s = 0; s < 8; ++s) {
            v4h t0;
#pragma unroll
            for (int i = 0; i < 4; ++i) t0[i] = (_Float16)src[s * 4 + i];
            *(v4h*)&cF[dp][n0 + kperm(s * 4)] = t0;
        }
    }
    __syncthreads();

    // GEMM: 128 rows x 128 cols, K = 64. Wave -> row block, 8 col blocks.
    v8f acc[8] = {};
#pragma unroll
    for (int kk = 0; kk < NDIM; kk += 32) {
        v16h a = *(const v16h*)&hT[wave * 16 + (lane & 15)][kk + (lane >> 4) * 16];
#pragma unroll
        for (int cb = 0; cb < 8; ++cb) {
            v16h b = *(const v16h*)&cF[cb * 16 + (lane & 15)][kk + (lane >> 4) * 16];
            acc[cb] = __builtin_amdgcn_wmma_f32_16x16x32_f16(
                false, a, false, b, (short)0, acc[cb], false, false);
        }
    }

    // direct coalesced stores from D layout
    {
        int moff = (lane >> 4) << 3;
        int ncol = lane & 15;
#pragma unroll
        for (int cb = 0; cb < 8; ++cb)
#pragma unroll
            for (int v = 0; v < 8; ++v) {
                int m = row0 + wave * 16 + moff + v;
                int d = d0 + cb * 16 + ncol;
                y[(size_t)m * DDIM + d] = acc[cb][v];
            }
    }
}

extern "C" void kernel_launch(void* const* d_in, const int* in_sizes, int n_in,
                              void* d_out, int out_size, void* d_ws, size_t ws_size,
                              hipStream_t stream) {
    (void)in_sizes; (void)n_in; (void)out_size; (void)ws_size;
    const float* x          = (const float*)d_in[0];
    const float* log_lambda = (const float*)d_in[1];
    const float* Bm         = (const float*)d_in[2];
    const float* Cm         = (const float*)d_in[3];
    const float* log_dt     = (const float*)d_in[4];
    float* y = (float*)d_out;

    float* hs      = (float*)d_ws;               // [BT][N]   8 MB
    float* Scarry  = hs + (size_t)BT * NDIM;     // [NCH][N]  64 KB
    float* carryIn = Scarry + (size_t)NCH * NDIM;// [NCH][N]  64 KB

    gemm1_scan<<<NCH, 256, 0, stream>>>(x, Bm, log_lambda, log_dt, hs, Scarry);
    carry_kernel<<<1, 512, 0, stream>>>(Scarry, log_lambda, log_dt, carryIn);
    fixup_gemm2<<<dim3(NCH, 4), 256, 0, stream>>>(hs, carryIn, Cm, log_lambda, log_dt, y);
}
